// ASGNet_41180146434125
// MI455X (gfx1250) — matrix operand order repaired and checked by reading, not compiled, not run on hardware
//
#include <hip/hip_runtime.h>
#include <hip/hip_bf16.h>

#define B_    4
#define N_    2048
#define D_    256
#define H_    8
#define HD_   32
#define BN_   (B_*N_)        // 8192 rows
#define ROWT_ (BN_/16)       // 512 row tiles
#define NT_   (N_/16)        // 128 key tiles per (b,h)
#define KT_   (D_/32)        // 8 K-steps for D=256
#define QCT_  (3*D_/16)      // 48 col tiles for QKV weights

typedef __bf16 bf16_t;
typedef __attribute__((ext_vector_type(16))) __bf16 v16bf;
typedef __attribute__((ext_vector_type(8)))  float  v8f;
typedef __attribute__((ext_vector_type(4))) unsigned int u32x4;
typedef __attribute__((ext_vector_type(8))) int i32x8;
typedef __attribute__((ext_vector_type(4))) int i32x4;

__device__ __forceinline__ v8f wmma_bf16(v16bf a, v16bf b, v8f c) {
  // D = A(16x32 bf16) x B(32x16 bf16) + C(16x16 f32)
  return __builtin_amdgcn_wmma_f32_16x16x32_bf16(false, a, false, b, (short)0, c, false, false);
}

// WMMA operand lane maps used here (contiguous k, same for A and B; not the ISA table order):
// A 16x32 bf16: lanes0-15 hold M=lane, K 0-15 (elem 0-15); lanes16-31: K 16-31
__device__ __forceinline__ int a_lane(int m, int k) { return m | ((k >> 4) << 4); }
__device__ __forceinline__ int a_elem(int k)        { return k & 15; }
// B 32x16 bf16: lanes0-15 N=lane hold K 0-15, lanes16-31 hold K 16-31
__device__ __forceinline__ int b_lane(int n, int k) { return n | ((k >> 4) << 4); }
__device__ __forceinline__ int b_elem(int k)        { return k & 15; }

// order-preserving float<->uint map for top-k threshold search
__device__ __forceinline__ float u2f(unsigned u) {
  unsigned b = (u & 0x80000000u) ? (u & 0x7fffffffu) : ~u;
  return __uint_as_float(b);
}

// ---- Tensor Data Mover: async 2D tile (16x16 x 4B elements) Global -> LDS ----
// D# per CDNA5 ISA: group0 = count/lds_addr/global_addr/type=2,
// group1 = data_size=4B, tensor_dim0/1, tile 16x16, dim0 stride. Groups 2/3 unused (2D).
__device__ __forceinline__ void tdm_load_tile16x16_b32(unsigned lds_addr, const void* gptr,
                                                       unsigned tdim0, unsigned tdim1,
                                                       unsigned stride0) {
  unsigned long long ga = (unsigned long long)gptr;
  u32x4 g0;
  g0[0] = 1u;                                            // count=1, user mode
  g0[1] = lds_addr;                                      // LDS byte address
  g0[2] = (unsigned)ga;                                  // global_addr[31:0]
  g0[3] = (unsigned)((ga >> 32) & 0x1ffffffu) | (2u << 30); // global_addr[56:32], type=2
  i32x8 g1;
  g1[0] = (int)(2u << 16);                               // workgroup_mask=0, data_size=4B
  g1[1] = (int)((tdim0 & 0xffffu) << 16);                // tensor_dim0[15:0]
  g1[2] = (int)((tdim0 >> 16) | ((tdim1 & 0xffffu) << 16)); // dim0[31:16], dim1[15:0]
  g1[3] = (int)((tdim1 >> 16) | (16u << 16));            // dim1[31:16], tile_dim0=16
  g1[4] = 16;                                            // tile_dim1=16, tile_dim2=0
  g1[5] = (int)stride0;                                  // tensor_dim0_stride[31:0]
  g1[6] = 0;                                             // stride hi, dim1_stride lo
  g1[7] = 0;
  i32x4 z4 = {0, 0, 0, 0};
#if __clang_major__ >= 23
  i32x8 z8 = {0, 0, 0, 0, 0, 0, 0, 0};
  __builtin_amdgcn_tensor_load_to_lds(g0, g1, z4, z4, z8, 0);
#else
  __builtin_amdgcn_tensor_load_to_lds(g0, g1, z4, z4, 0);
#endif
}

// ---------------- init reg-loss accumulator ----------------
__global__ void k_init(float* accum) { if (threadIdx.x == 0) accum[0] = 0.0f; }

// ---------------- convert weights to bf16, pre-swizzled B-operand layout ----------------
__global__ void k_wcvt(const float* __restrict__ qkv_w, const float* __restrict__ out_w,
                       bf16_t* __restrict__ wq_sw, bf16_t* __restrict__ ow_sw) {
  const int t1 = 3 * D_ * D_, t2 = D_ * D_;
  for (int idx = blockIdx.x * blockDim.x + threadIdx.x; idx < t1 + t2;
       idx += gridDim.x * blockDim.x) {
    if (idx < t1) {
      int r = idx / D_, c = idx % D_;   // B[k=c, n=r]: C = xn @ W^T
      int o = ((r >> 4) * KT_ + (c >> 5)) * 512 + b_lane(r & 15, c & 31) * 16 + b_elem(c & 31);
      wq_sw[o] = (bf16_t)qkv_w[idx];
    } else {
      int j = idx - t1;
      int r = j / D_, c = j % D_;
      int o = ((r >> 4) * KT_ + (c >> 5)) * 512 + b_lane(r & 15, c & 31) * 16 + b_elem(c & 31);
      ow_sw[o] = (bf16_t)out_w[j];
    }
  }
}

// ---------------- LayerNorm, write xn as bf16 in A-operand swizzle ----------------
__global__ void k_ln(const float* __restrict__ x, const float* __restrict__ g,
                     const float* __restrict__ bta, bf16_t* __restrict__ xn_sw) {
  __shared__ float red[256];
  int row = blockIdx.x, t = threadIdx.x;
  float v = x[row * D_ + t];
  red[t] = v; __syncthreads();
  for (int s = 128; s; s >>= 1) { if (t < s) red[t] += red[t + s]; __syncthreads(); }
  float mu = red[0] * (1.0f / D_);
  __syncthreads();
  float d = v - mu;
  red[t] = d * d; __syncthreads();
  for (int s = 128; s; s >>= 1) { if (t < s) red[t] += red[t + s]; __syncthreads(); }
  float var = red[0] * (1.0f / D_);
  float xn = d * rsqrtf(var + 1e-5f) * g[t] + bta[t];
  int kt = t >> 5, kp = t & 31;
  xn_sw[(((row >> 4) * KT_ + kt) * 32 + a_lane(row & 15, kp)) * 16 + a_elem(kp)] = (bf16_t)xn;
}

// ---------------- QKV GEMM: [8192,256] @ [256,768] via WMMA; scatter to swizzled q/k/v ----------------
__global__ void k_qkv(const bf16_t* __restrict__ xn_sw, const bf16_t* __restrict__ wq_sw,
                      const float* __restrict__ qkv_b,
                      bf16_t* __restrict__ q_sw, bf16_t* __restrict__ k_sw,
                      bf16_t* __restrict__ v_sw) {
  int lane = threadIdx.x & 31, wave = threadIdx.x >> 5;
  int rowtile = blockIdx.x;
  int coltile = blockIdx.y * 8 + wave;
  v8f c = {};
#pragma unroll
  for (int kt = 0; kt < KT_; ++kt) {
    v16bf a = *(const v16bf*)(xn_sw + ((rowtile * KT_ + kt) * 32 + lane) * 16);
    v16bf b = *(const v16bf*)(wq_sw + ((coltile * KT_ + kt) * 32 + lane) * 16);
    c = wmma_bf16(a, b, c);
  }
  int j = coltile * 16 + (lane & 15);
  int which = j >> 8, h = (j >> 5) & 7, hd = j & 31;
  float bias = qkv_b[j];
#pragma unroll
  for (int r = 0; r < 8; ++r) {
    int m = r + ((lane >> 4) << 3);
    int grow = rowtile * 16 + m;
    int b_ = grow >> 11, nn = grow & (N_ - 1);
    int bh = b_ * H_ + h;
    bf16_t bv = (bf16_t)(c[r] + bias);
    if (which == 0) {        // q: A-operand layout per (b,h,row-tile)
      q_sw[((bh * NT_ + (nn >> 4)) * 32 + a_lane(nn & 15, hd)) * 16 + a_elem(hd)] = bv;
    } else if (which == 1) { // k: B-operand layout (K=hd, N=key index)
      k_sw[((bh * NT_ + (nn >> 4)) * 32 + b_lane(nn & 15, hd)) * 16 + b_elem(hd)] = bv;
    } else {                 // v: B-operand layout (K=key index chunk of 32, N=hd half of 16)
      v_sw[(((bh * 64 + (nn >> 5)) * 2 + (hd >> 4)) * 32 + b_lane(hd & 15, nn & 31)) * 16 +
           b_elem(nn & 31)] = bv;
    }
  }
}

// ---------------- attention: TDM-staged bias/adj, scores WMMA -> LDS, top-k, softmax, attn@v ----------------
__global__ void k_attn(const bf16_t* __restrict__ q_sw, const bf16_t* __restrict__ k_sw,
                       const bf16_t* __restrict__ v_sw,
                       const float* __restrict__ att_bias, const int* __restrict__ adj,
                       const float* __restrict__ adj_scale, const float* __restrict__ sparth,
                       bf16_t* __restrict__ att_sw, float* __restrict__ accum) {
  extern __shared__ char smem[];
  float*  scores = (float*)smem;                                  // 16 x 2048 f32 = 128KB
  bf16_t* attnA  = (bf16_t*)(smem + 16 * N_ * 4);                 // 64 chunks x 32 x 16 bf16 = 64KB
  float*  accf   = (float*)(smem + 16 * N_ * 4 + 64 * 512 * 2);   // 16 x 32 f32 = 2KB
  float*  biasst = (float*)(smem + 198656);                       // 8 waves x 16x16 f32 = 8KB
  int*    adjst  = (int*)(smem + 198656 + 8192);                  // 8 waves x 16x16 i32 = 8KB

  int lane = threadIdx.x & 31, wave = threadIdx.x >> 5;
  int ntile = blockIdx.x & (NT_ - 1);
  int bh = blockIdx.x >> 7;
  int h = bh & 7;
  float scale = adj_scale[0];
  const float rs = 0.17677669529663687f; // 1/sqrt(HD)

  float* bias_buf = biasst + wave * 256;
  int*   adj_buf  = adjst + wave * 256;
  unsigned bias_lds = (unsigned)(unsigned long long)(void*)bias_buf; // LDS byte addr = generic[31:0]
  unsigned adj_lds  = (unsigned)(unsigned long long)(void*)adj_buf;

  // phase 1: one WMMA per 16x16 score tile (K = HD = 32 exactly);
  // bias/adj 16x16 strided tiles DMA'd by the Tensor Data Mover, overlapped with the WMMA path
  v16bf aq = *(const v16bf*)(q_sw + ((bh * NT_ + ntile) * 32 + lane) * 16);
  for (int j = wave; j < NT_; j += 8) {
    // async: TDM pulls this tile's bias + adj columns into per-wave LDS staging
    tdm_load_tile16x16_b32(bias_lds, att_bias + (h * N_ + ntile * 16) * N_ + j * 16,
                           N_, H_ * N_, N_);
    tdm_load_tile16x16_b32(adj_lds, adj + (ntile * 16) * N_ + j * 16, N_, N_, N_);
    v16bf kb = *(const v16bf*)(k_sw + ((bh * NT_ + j) * 32 + lane) * 16);
    v8f c = {};
    c = wmma_bf16(aq, kb, c);                 // overlaps with outstanding TENSORcnt ops
    __builtin_amdgcn_s_wait_tensorcnt(0);     // tiles landed in LDS
    int m = j * 16 + (lane & 15);
    int tx = lane & 15;
#pragma unroll
    for (int r = 0; r < 8; ++r) {
      int M = r + ((lane >> 4) << 3);
      float s = c[r] * rs + bias_buf[M * 16 + tx];
      s += scale * ((adj_buf[M * 16 + tx] > 0) ? 0.0f : -1e9f);
      scores[M * N_ + m] = s;
    }
    // single-buffer staging: drain LDS reads before next iteration's TDM overwrites it
    asm volatile("s_wait_dscnt 0x0" ::: "memory");
  }
  __syncthreads();

  // phase 2: per-row top-k via bitwise threshold search on order-preserving uint, then softmax
  {
    float t = 1.0f / (1.0f + expf(-sparth[h]));
    int kk = (int)(N_ * (1.0f - t));
    if (kk < 1) kk = 1;
    for (int rr = 0; rr < 2; ++rr) {
      int row = wave * 2 + rr;
      unsigned uv[64];
      unsigned umax = 0u;
#pragma unroll
      for (int i = 0; i < 64; ++i) {
        unsigned bb = __float_as_uint(scores[row * N_ + i * 32 + lane]);
        unsigned u = (bb & 0x80000000u) ? ~bb : (bb | 0x80000000u);
        uv[i] = u; if (u > umax) umax = u;
      }
      for (int off = 16; off; off >>= 1) {
        unsigned o = (unsigned)__shfl_xor((int)umax, off, 32);
        if (o > umax) umax = o;
      }
      unsigned thr = 0u;
      for (int bit = 31; bit >= 0; --bit) {     // find k-th largest value
        unsigned cand = thr | (1u << bit);
        int cnt = 0;
#pragma unroll
        for (int i = 0; i < 64; ++i) cnt += (uv[i] >= cand) ? 1 : 0;
        for (int off = 16; off; off >>= 1) cnt += __shfl_xor(cnt, off, 32);
        if (cnt >= kk) thr = cand;
      }
      float fmx = u2f(umax);
      float lsum = 0.0f;
#pragma unroll
      for (int i = 0; i < 64; ++i)
        if (uv[i] >= thr) lsum += expf(u2f(uv[i]) - fmx);
      for (int off = 16; off; off >>= 1) lsum += __shfl_xor(lsum, off, 32);
      float inv = 1.0f / lsum;
      float psum = 0.0f;
#pragma unroll
      for (int i = 0; i < 64; ++i) {            // write probs into A-operand swizzle in LDS
        int cidx = i * 32 + lane;
        float p = (uv[i] >= thr) ? expf(u2f(uv[i]) - fmx) * inv : 0.0f;
        psum += p;
        int kp = cidx & 31, chunk = cidx >> 5;
        attnA[(chunk * 32 + a_lane(row, kp)) * 16 + a_elem(kp)] = (bf16_t)p;
      }
      for (int off = 16; off; off >>= 1) psum += __shfl_xor(psum, off, 32);
      if (lane == 0) atomicAdd(accum, psum);    // for reg loss
    }
  }
  __syncthreads();

  // phase 3: attn @ v; each wave owns 8 K-chunks, cross-wave reduce via ds_add_f32
  for (int i = threadIdx.x; i < 512; i += 256) accf[i] = 0.0f;
  __syncthreads();
  v8f c0 = {}, c1 = {};
#pragma unroll
  for (int cc = 0; cc < 8; ++cc) {
    int chunk = wave * 8 + cc;
    v16bf a  = *(const v16bf*)(attnA + (chunk * 32 + lane) * 16);
    v16bf b0 = *(const v16bf*)(v_sw + (((bh * 64 + chunk) * 2 + 0) * 32 + lane) * 16);
    v16bf b1 = *(const v16bf*)(v_sw + (((bh * 64 + chunk) * 2 + 1) * 32 + lane) * 16);
    c0 = wmma_bf16(a, b0, c0);
    c1 = wmma_bf16(a, b1, c1);
  }
#pragma unroll
  for (int r = 0; r < 8; ++r) {
    int M = r + ((lane >> 4) << 3);
    atomicAdd(&accf[M * 32 + (lane & 15)], c0[r]);
    atomicAdd(&accf[M * 32 + 16 + (lane & 15)], c1[r]);
  }
  __syncthreads();

  // write attended, pre-swizzled as A operand for the output projection
  int b_ = bh >> 3;
  for (int i = threadIdx.x; i < 512; i += 256) {
    int M = i >> 5, hd = i & 31;
    int grow = b_ * N_ + ntile * 16 + M;
    int rowtile = grow >> 4;
    att_sw[((rowtile * KT_ + h) * 32 + a_lane(M, hd)) * 16 + a_elem(hd)] = (bf16_t)accf[i];
  }
}

// ---------------- output projection + bias + residual ----------------
__global__ void k_outp(const bf16_t* __restrict__ att_sw, const bf16_t* __restrict__ ow_sw,
                       const float* __restrict__ out_b, const float* __restrict__ x,
                       float* __restrict__ out) {
  int lane = threadIdx.x & 31, wave = threadIdx.x >> 5;
  int rowtile = blockIdx.x;
#pragma unroll
  for (int cw = 0; cw < 2; ++cw) {
    int coltile = wave + cw * 8;
    v8f c = {};
#pragma unroll
    for (int kt = 0; kt < KT_; ++kt) {
      v16bf a = *(const v16bf*)(att_sw + ((rowtile * KT_ + kt) * 32 + lane) * 16);
      v16bf b = *(const v16bf*)(ow_sw + ((coltile * KT_ + kt) * 32 + lane) * 16);
      c = wmma_bf16(a, b, c);
    }
    int dcol = coltile * 16 + (lane & 15);
    float ob = out_b[dcol];
#pragma unroll
    for (int r = 0; r < 8; ++r) {
      int grow = rowtile * 16 + r + ((lane >> 4) << 3);
      out[grow * D_ + dcol] = c[r] + ob + x[grow * D_ + dcol];
    }
  }
}

// ---------------- reg loss ----------------
__global__ void k_fin(const float* __restrict__ accum, const float* __restrict__ l1,
                      float* __restrict__ out) {
  if (threadIdx.x == 0) {
    float sp = log1pf(expf(l1[0]));
    out[BN_ * D_] = sp * accum[0] / ((float)B_ * H_ * N_ * N_);
  }
}

extern "C" void kernel_launch(void* const* d_in, const int* in_sizes, int n_in,
                              void* d_out, int out_size, void* d_ws, size_t ws_size,
                              hipStream_t stream) {
  (void)in_sizes; (void)n_in; (void)out_size; (void)ws_size;
  const float* x        = (const float*)d_in[0];
  const int*   adj      = (const int*)d_in[1];
  const float* qkv_w    = (const float*)d_in[2];
  const float* qkv_b    = (const float*)d_in[3];
  const float* out_w    = (const float*)d_in[4];
  const float* out_b    = (const float*)d_in[5];
  const float* ln_g     = (const float*)d_in[6];
  const float* ln_b     = (const float*)d_in[7];
  const float* att_bias = (const float*)d_in[8];
  const float* adj_s    = (const float*)d_in[9];
  const float* sparth   = (const float*)d_in[10];
  const float* l1       = (const float*)d_in[11];
  float* out = (float*)d_out;

  char* ws = (char*)d_ws;
  size_t off = 0;
  bf16_t* xn_sw  = (bf16_t*)(ws + off); off += (size_t)ROWT_ * KT_ * 512 * 2;     // 4 MB
  bf16_t* wq_sw  = (bf16_t*)(ws + off); off += (size_t)QCT_ * KT_ * 512 * 2;      // 384 KB
  bf16_t* ow_sw  = (bf16_t*)(ws + off); off += (size_t)16 * KT_ * 512 * 2;        // 128 KB
  bf16_t* q_sw   = (bf16_t*)(ws + off); off += (size_t)B_ * H_ * NT_ * 512 * 2;   // 4 MB
  bf16_t* k_sw   = (bf16_t*)(ws + off); off += (size_t)B_ * H_ * NT_ * 512 * 2;   // 4 MB
  bf16_t* v_sw   = (bf16_t*)(ws + off); off += (size_t)B_ * H_ * 128 * 512 * 2;   // 4 MB
  bf16_t* att_sw = (bf16_t*)(ws + off); off += (size_t)ROWT_ * KT_ * 512 * 2;     // 4 MB
  float*  accum  = (float*)(ws + off);  off += 256;

  k_init<<<1, 32, 0, stream>>>(accum);
  k_wcvt<<<256, 256, 0, stream>>>(qkv_w, out_w, wq_sw, ow_sw);
  k_ln<<<BN_, 256, 0, stream>>>(x, ln_g, ln_b, xn_sw);
  k_qkv<<<dim3(ROWT_, 6), 256, 0, stream>>>(xn_sw, wq_sw, qkv_b, q_sw, k_sw, v_sw);
  // LDS: scores 128K + attnA 64K + accf 2K + bias/adj staging 16K = 215040 B < 320KB WGP limit
  size_t shmem = 198656 + 8192 + 8192;
  k_attn<<<B_ * H_ * NT_, 256, shmem, stream>>>(q_sw, k_sw, v_sw, att_bias, adj, adj_s,
                                                sparth, att_sw, accum);
  k_outp<<<ROWT_, 256, 0, stream>>>(att_sw, ow_sw, out_b, x, out);
  k_fin<<<1, 32, 0, stream>>>(accum, l1, out);
}
